// LSTM_3994319586039
// MI455X (gfx1250) — matrix operand order repaired and checked by reading, not compiled, not run on hardware
//
#include <hip/hip_runtime.h>

typedef __attribute__((ext_vector_type(16))) _Float16 v16h;
typedef __attribute__((ext_vector_type(8)))  float    v8f;

#define BB    256
#define TT    512
#define DIN   128
#define HH    512
#define NWG   16      // workgroups in lstm kernel (each owns 32 hidden units)
#define KCH   16      // k-chunks covering H (512/32)

// --- A-fragment layout (16-bit A 16x32, ISA 7.12.2):
// lane 0-15: M=lane, halves 0..7 -> K=0..7, halves 8..15 -> K=16..23
// lane 16-31: M=lane-16, K offset +8
__device__ __forceinline__ int a_klocal(int p, int lane) {
    return ((p < 4) ? (2 * p) : (16 + 2 * (p - 4))) + ((lane & 16) ? 8 : 0);
}

__device__ __forceinline__ float fsig(float x) {
    return __builtin_amdgcn_rcpf(1.f + __builtin_amdgcn_exp2f(-1.4426950408889634f * x));
}
__device__ __forceinline__ float ftanhf(float x) {
#if __has_builtin(__builtin_amdgcn_tanhf)
    return __builtin_amdgcn_tanhf(x);
#else
    return 1.f - 2.f * __builtin_amdgcn_rcpf(1.f + __builtin_amdgcn_exp2f(2.8853900817779268f * x));
#endif
}

__global__ void k_init(unsigned* ctrs) {
    if (threadIdx.x < 64) ctrs[threadIdx.x] = 0u;
}

// x [B,T,128] f32 -> A-fragments [T][16][4][32][8 u32] f16-pairs
__global__ void k_pack_x(const float* __restrict__ x, unsigned* __restrict__ xf) {
    unsigned idx = blockIdx.x * blockDim.x + threadIdx.x;   // TT*16*4*32*8 = 8388608
    int p    = idx & 7;
    int lane = (idx >> 3) & 31;
    int kc   = (idx >> 8) & 3;
    int mt   = (idx >> 10) & 15;
    int t    = idx >> 14;
    int b    = mt * 16 + (lane & 15);
    int k    = kc * 32 + a_klocal(p, lane);
    const float* s = x + ((size_t)b * TT + t) * DIN + k;
    union { unsigned u; _Float16 h[2]; } cv;
    cv.h[0] = (_Float16)s[0];
    cv.h[1] = (_Float16)s[1];
    xf[idx] = cv.u;
}

// Wcat = [W_ih | W_hh] packed into B-fragments, wg-slice major:
// [wg(16)][tile(8)=half*4+gate][kc(KC)][lane(32)][8 u32]
// B layout (16-bit 32x16): lanes 0-15 N=lane K=0..7,16..23; lanes 16-31 N=lane-16 K=8..15,24..31
__global__ void k_pack_w(const float* __restrict__ Wih, const float* __restrict__ Whh,
                         unsigned* __restrict__ wp, int dIn, int KC, int total) {
    int idx = blockIdx.x * blockDim.x + threadIdx.x;
    if (idx >= total) return;
    int p    = idx & 7;
    int lane = (idx >> 3) & 31;
    int rest = idx >> 8;
    int kc   = rest % KC; rest /= KC;
    int tile = rest & 7;
    int wg   = rest >> 3;
    int gate = tile & 3, half = tile >> 2;
    int col  = gate * HH + wg * 32 + half * 16 + (lane & 15);
    int k    = kc * 32 + a_klocal(p, lane);
    float v0, v1;
    if (k < dIn) { v0 = Wih[(size_t)col * dIn + k];        v1 = Wih[(size_t)col * dIn + k + 1]; }
    else         { v0 = Whh[(size_t)col * HH + (k - dIn)]; v1 = Whh[(size_t)col * HH + (k - dIn) + 1]; }
    union { unsigned u; _Float16 h[2]; } cv;
    cv.h[0] = (_Float16)v0;
    cv.h[1] = (_Float16)v1;
    wp[idx] = cv.u;
}

// Persistent recurrent kernel. 16 WGs x 512 threads (16 waves).
// Wave = row-tile mt; owns 8 acc tiles = 4 gates x 2 halves for units [wg*32, wg*32+32).
// Weight slice resident in LDS for all 512 steps. h exchanged via global fragments +
// device-scope barrier each step.
template<int KCX>
__global__ __launch_bounds__(512, 1)
void k_lstm(const unsigned* __restrict__ Xf,     // input fragments [T][16][KCX][32][8]
            const _Float16* __restrict__ Wp,     // packed weights (this layer, all wg slices)
            const float* __restrict__ bih, const float* __restrict__ bhh,
            unsigned* Hf,                        // output fragments [T][16][KCH][32][8]
            unsigned* barrier_ctr)
{
    constexpr int KC = KCX + KCH;
    __shared__ _Float16 Wlds[8 * KC * 32 * 16];       // 160KB (l0) / 256KB (l1,l2)
    __shared__ _Float16 hstage[16 * 16 * 32];         // 16KB transpose bounce

    const int tid  = threadIdx.x;
    const int lane = tid & 31;
    const int mt   = tid >> 5;          // row-tile 0..15
    const int wg   = blockIdx.x;        // unit-slice 0..15

    // stage weight slice into LDS (read from HBM/L2 exactly once)
    {
        const uint4* src = (const uint4*)(Wp + (size_t)wg * (8 * KC * 32 * 16));
        uint4* dst = (uint4*)Wlds;
        const int n = 8 * KC * 32 * 2;  // uint4 count
        for (int i = tid; i < n; i += 512) dst[i] = src[i];
    }

    // per-lane bias (depends only on N=lane&15), tile j = half*4 + gate
    float bias[8];
#pragma unroll
    for (int j = 0; j < 8; ++j) {
        int gate = j & 3, half = j >> 2;
        int col  = gate * HH + wg * 32 + half * 16 + (lane & 15);
        bias[j] = bih[col] + bhh[col];
    }

    v8f cst[2];                          // f32 cell state, C-matrix layout
#pragma unroll
    for (int hf2 = 0; hf2 < 2; ++hf2)
#pragma unroll
        for (int r = 0; r < 8; ++r) cst[hf2][r] = 0.f;

    __syncthreads();

    for (int t = 0; t < TT; ++t) {
        v8f acc[8];
#pragma unroll
        for (int j = 0; j < 8; ++j) {
            v8f a;
#pragma unroll
            for (int r = 0; r < 8; ++r) a[r] = bias[j];
            acc[j] = a;
        }

        const int kcEnd = (t == 0) ? KCX : KC;   // no h contribution at t==0
        for (int kc = 0; kc < kcEnd; ++kc) {
            union { uint4 q[2]; v16h v; } A;
            const uint4* ap;
            if (kc < KCX)
                ap = (const uint4*)(Xf + (((size_t)t * 16 + mt) * KCX + kc) * 256 + lane * 8);
            else
                ap = (const uint4*)(Hf + (((size_t)(t - 1) * 16 + mt) * KCH + (kc - KCX)) * 256 + lane * 8);
            A.q[0] = ap[0];
            A.q[1] = ap[1];
#pragma unroll
            for (int j = 0; j < 8; ++j) {
                const v16h b = *(const v16h*)&Wlds[(((size_t)j * KC + kc) * 32 + lane) * 16];
                acc[j] = __builtin_amdgcn_wmma_f32_16x16x32_f16(
                    false, A.v, false, b, (short)0, acc[j], false, false);
            }
        }

        // Warm the WGP cache with next step's x fragments while we do the
        // gate math / barrier (lowers to global_prefetch_b8; h fragments for
        // t+1 are being produced elsewhere and cannot be prefetched).
        if (t + 1 < TT) {
#pragma unroll
            for (int kc = 0; kc < KCX; ++kc)
                __builtin_prefetch(
                    Xf + (((size_t)(t + 1) * 16 + mt) * KCX + kc) * 256 + lane * 8, 0, 3);
        }

        // gates (PyTorch order i,f,g,o) + cell/hidden update, all in C layout
        _Float16 hval[2][8];
#pragma unroll
        for (int half = 0; half < 2; ++half) {
            v8f ai = acc[half * 4 + 0], af = acc[half * 4 + 1];
            v8f ag = acc[half * 4 + 2], ao = acc[half * 4 + 3];
#pragma unroll
            for (int r = 0; r < 8; ++r) {
                float iv = fsig(ai[r]);
                float fv = fsig(af[r]);
                float gv = ftanhf(ag[r]);
                float ov = fsig(ao[r]);
                float c  = fv * cst[half][r] + iv * gv;
                cst[half][r] = c;
                hval[half][r] = (_Float16)(ov * ftanhf(c));
            }
        }

        // C-layout -> A-fragment transpose via LDS bounce
#pragma unroll
        for (int half = 0; half < 2; ++half)
#pragma unroll
            for (int r = 0; r < 8; ++r) {
                int row = r + ((lane & 16) ? 8 : 0);
                hstage[(mt * 16 + row) * 32 + half * 16 + (lane & 15)] = hval[half][r];
            }
        __syncthreads();

        union { uint4 q[2]; unsigned u[8]; } O;
#pragma unroll
        for (int p = 0; p < 8; ++p) {
            int kl = a_klocal(p, lane);   // even -> 4B aligned pair
            O.u[p] = *(const unsigned*)&hstage[(mt * 16 + (lane & 15)) * 32 + kl];
        }
        uint4* op = (uint4*)(Hf + (((size_t)t * 16 + mt) * KCH + wg) * 256 + lane * 8);
        op[0] = O.q[0];
        op[1] = O.q[1];

        if (t == TT - 1) break;

        // device-scope step barrier across the 16 co-resident WGs
        __threadfence();
        __syncthreads();
        if (tid == 0) {
            __hip_atomic_fetch_add(barrier_ctr, 1u, __ATOMIC_RELEASE, __HIP_MEMORY_SCOPE_AGENT);
            unsigned target = (unsigned)(NWG * (t + 1));
            while (__hip_atomic_load(barrier_ctr, __ATOMIC_ACQUIRE, __HIP_MEMORY_SCOPE_AGENT) < target)
                __builtin_amdgcn_s_sleep(1);
        }
        __syncthreads();
    }
}

// out[b] = h_last[b,:] . W_fc[0,:] + b_fc   (D_OUT = 1)
__global__ void k_fc(const unsigned* __restrict__ Hf, const float* __restrict__ Wfc,
                     const float* __restrict__ bfc, float* __restrict__ out) {
    __shared__ float red[16];
    int tid  = threadIdx.x;        // 256
    int lane = tid & 31;
    int p    = tid >> 5;           // 0..7
    int mt   = blockIdx.x;         // 0..15
    if (tid < 16) red[tid] = 0.f;
    __syncthreads();
    int base = a_klocal(p, lane);
    float sum = 0.f;
    for (int kc = 0; kc < KCH; ++kc) {
        unsigned u = Hf[(((size_t)(TT - 1) * 16 + mt) * KCH + kc) * 256 + lane * 8 + p];
        union { unsigned u; _Float16 h[2]; } cv; cv.u = u;
        int k = kc * 32 + base;
        sum += (float)cv.h[0] * Wfc[k] + (float)cv.h[1] * Wfc[k + 1];
    }
    atomicAdd(&red[lane & 15], sum);
    __syncthreads();
    if (tid < 16) out[mt * 16 + tid] = red[tid] + bfc[0];
}

extern "C" void kernel_launch(void* const* d_in, const int* in_sizes, int n_in,
                              void* d_out, int out_size, void* d_ws, size_t ws_size,
                              hipStream_t stream) {
    const float* x      = (const float*)d_in[0];
    const float* Wih[3] = {(const float*)d_in[1], (const float*)d_in[5], (const float*)d_in[9]};
    const float* Whh[3] = {(const float*)d_in[2], (const float*)d_in[6], (const float*)d_in[10]};
    const float* bih[3] = {(const float*)d_in[3], (const float*)d_in[7], (const float*)d_in[11]};
    const float* bhh[3] = {(const float*)d_in[4], (const float*)d_in[8], (const float*)d_in[12]};
    const float* Wfc    = (const float*)d_in[13];
    const float* bfc    = (const float*)d_in[14];
    float* out = (float*)d_out;

    char* ws = (char*)d_ws;
    size_t off = 0;
    auto take = [&](size_t bytes) -> char* {
        char* p = ws + off;
        off += (bytes + 255) & ~(size_t)255;
        return p;
    };
    unsigned* ctrs = (unsigned*)take(256);
    unsigned* xf   = (unsigned*)take((size_t)TT * 16 * 4 * 32 * 8 * 4);     //  32 MB
    unsigned* hA   = (unsigned*)take((size_t)TT * 16 * KCH * 32 * 8 * 4);   // 128 MB
    unsigned* hB   = (unsigned*)take((size_t)TT * 16 * KCH * 32 * 8 * 4);   // 128 MB
    unsigned* wp0  = (unsigned*)take((size_t)16 * 8 * 20 * 32 * 8 * 4);
    unsigned* wp1  = (unsigned*)take((size_t)16 * 8 * 32 * 32 * 8 * 4);
    unsigned* wp2  = (unsigned*)take((size_t)16 * 8 * 32 * 32 * 8 * 4);

    k_init<<<1, 64, 0, stream>>>(ctrs);
    k_pack_x<<<(TT * 16 * 4 * 32 * 8) / 256, 256, 0, stream>>>(x, xf);

    int tot0 = 16 * 8 * 20 * 32 * 8;
    int tot  = 16 * 8 * 32 * 32 * 8;
    k_pack_w<<<(tot0 + 255) / 256, 256, 0, stream>>>(Wih[0], Whh[0], wp0, DIN, 20, tot0);
    k_pack_w<<<(tot  + 255) / 256, 256, 0, stream>>>(Wih[1], Whh[1], wp1, HH,  32, tot);
    k_pack_w<<<(tot  + 255) / 256, 256, 0, stream>>>(Wih[2], Whh[2], wp2, HH,  32, tot);

    k_lstm<4><<<NWG, 512, 0, stream>>>(xf, (const _Float16*)wp0, bih[0], bhh[0], hA, ctrs + 0);
    k_lstm<16><<<NWG, 512, 0, stream>>>(hA, (const _Float16*)wp1, bih[1], bhh[1], hB, ctrs + 8);
    k_lstm<16><<<NWG, 512, 0, stream>>>(hB, (const _Float16*)wp2, bih[2], bhh[2], hA, ctrs + 16);

    k_fc<<<16, 256, 0, stream>>>(hA, Wfc, bfc, out);
}
